// AttendRNN_14207751815691
// MI455X (gfx1250) — matrix operand order repaired and checked by reading, not compiled, not run on hardware
//
#include <hip/hip_runtime.h>
#include <hip/hip_bf16.h>

typedef __bf16 bf16;
typedef bf16  v16bf __attribute__((ext_vector_type(16)));
typedef float v8f   __attribute__((ext_vector_type(8)));

constexpr int kB    = 512;    // batch
constexpr int kN    = 200;    // sequence length
constexpr int kV    = 300;    // embed dim
constexpr int kH    = 300;    // hidden
constexpr int kG    = 900;    // 3*H
constexpr int kKP   = 320;    // K padded to mult of 32 (V and H)
constexpr int kRP   = 960;    // padded weight rows (>= max col-tile reach)
constexpr int kFCK  = 2400;   // fc1 K
constexpr int kFCH  = 512;    // fc1 N
constexpr int kO2W  = 600;    // 2*H
constexpr int kVoc  = 50000;  // vocab

#define WMMA_BF16(A, Bm, C) \
  __builtin_amdgcn_wmma_f32_16x16x32_bf16(false, (A), false, (Bm), (short)0, (C), false, false)

__device__ __forceinline__ v8f vzero() {
  v8f z = {0.f, 0.f, 0.f, 0.f, 0.f, 0.f, 0.f, 0.f};
  return z;
}

// A-matrix 16x32 bf16 fragment (row-major source, per-lane row pointer).
// elem j: K = k0 + half*16 + j
__device__ __forceinline__ v16bf frag_a(const bf16* __restrict__ row, int k0, int half) {
  const bf16* q = row + k0 + half * 16;
  v16bf f;
#pragma unroll
  for (int j = 0; j < 16; ++j) f[j] = q[j];
  return f;
}

// B-matrix 32x16 bf16 fragment where B[k][n] comes from a (N,K) row-major weight:
// per-lane pointer = W + n*ldk ; elem j: K = k0 + half*16 + j
__device__ __forceinline__ v16bf frag_b(const bf16* __restrict__ row, int k0, int half) {
  const bf16* q = row + k0 + half * 16;
  v16bf f;
#pragma unroll
  for (int j = 0; j < 16; ++j) f[j] = q[j];
  return f;
}

// B fragment with transposed access from LDS: B[k][n] = lds[k*ld + n]; base = lds + n
__device__ __forceinline__ v16bf frag_bT_lds(const bf16* __restrict__ base, int k0, int half, int ld) {
  int kk = k0 + half * 16;
  v16bf f;
#pragma unroll
  for (int j = 0; j < 16; ++j) f[j] = base[(kk + j) * ld];
  return f;
}

__device__ __forceinline__ void grid_barrier(int* bar, int nwg) {
  __syncthreads();
  if (threadIdx.x == 0) {
    __threadfence();
    int ticket = __hip_atomic_fetch_add(bar, 1, __ATOMIC_ACQ_REL, __HIP_MEMORY_SCOPE_AGENT);
    int target = (ticket / nwg + 1) * nwg;
    while (__hip_atomic_load(bar, __ATOMIC_ACQUIRE, __HIP_MEMORY_SCOPE_AGENT) < target)
      __builtin_amdgcn_s_sleep(2);
  }
  __syncthreads();
}

// ---------------- conversion f32 -> zero-padded bf16 ----------------
__global__ void cvt_pad_kernel(const float* __restrict__ src, bf16* __restrict__ dst,
                               int rows, int cols, int rowsP, int colsP) {
  long total = (long)rowsP * colsP;
  for (long i = blockIdx.x * (long)blockDim.x + threadIdx.x; i < total;
       i += (long)gridDim.x * blockDim.x) {
    int r = (int)(i / colsP), c = (int)(i % colsP);
    float v = (r < rows && c < cols) ? src[(long)r * cols + c] : 0.f;
    dst[i] = (bf16)v;
  }
}

__global__ void zero_kernel(unsigned int* __restrict__ p, long n) {
  for (long i = blockIdx.x * blockDim.x + threadIdx.x; i < n; i += (long)gridDim.x * blockDim.x)
    p[i] = 0u;
}

// ---------------- fused embedding-gather + input projection GEMM ----------------
// pre[b][n][g] = sum_v embBf[x[b,seq,n], v] * Wih[g, v] + bih[g]
// grid: (BN/64, ceil(G/64)=15, 2 dirs); 128 threads = 4 waves, wave = 16 rows x 64 cols
// B tile (64 x 320 bf16 = 40 KB) staged in LDS once: all 4 waves share identical B frags.
// A rows come straight from the L2-resident padded bf16 embedding table (clean b128 loads).
__global__ void proj_kernel(const int* __restrict__ x, const bf16* __restrict__ embBf,
                            const bf16* __restrict__ WihFp, const bf16* __restrict__ WihBp,
                            const float* __restrict__ bihF, const float* __restrict__ bihB,
                            bf16* __restrict__ preF, bf16* __restrict__ preB, int seq) {
  __shared__ bf16 ldsW[64 * kKP];
  const int lane = threadIdx.x & 31, wave = threadIdx.x >> 5;
  const int half = lane >> 4, l16 = lane & 15;
  const int m0 = blockIdx.x * 64 + wave * 16;
  const int n0 = blockIdx.y * 64;
  const int dir = blockIdx.z;
  const bf16* W = dir ? WihBp : WihFp;
  const float* bih = dir ? bihB : bihF;
  bf16* pre = dir ? preB : preF;

  // stage the 64-row weight tile (rows n0..n0+63 of the zero-padded (960x320) matrix)
  for (int i = threadIdx.x; i < 64 * (kKP / 2); i += 128) {
    int row = i / (kKP / 2), c2 = i % (kKP / 2);
    const unsigned int* src = (const unsigned int*)(W + (long)(n0 + row) * kKP);
    ((unsigned int*)ldsW)[i] = src[c2];
  }
  __syncthreads();

  int mrow = m0 + l16;               // flat (b,n) row, < B*N
  int b = mrow / kN, n = mrow - b * kN;
  int tok = x[(b * 2 + seq) * kN + n];
  const bf16* erow = embBf + (long)tok * kKP;

  v8f acc[4];
#pragma unroll
  for (int c = 0; c < 4; ++c) acc[c] = vzero();

  for (int kt = 0; kt < 10; ++kt) {
    int k0 = kt * 32;
    v16bf a = frag_a(erow, k0, half);
#pragma unroll
    for (int c = 0; c < 4; ++c) {
      acc[c] = WMMA_BF16(a, frag_b(ldsW + (long)(c * 16 + l16) * kKP, k0, half), acc[c]);
    }
  }
#pragma unroll
  for (int c = 0; c < 4; ++c) {
    int col = n0 + c * 16 + l16;
    if (col < kG) {
      float bias = bih[col];
#pragma unroll
      for (int r = 0; r < 8; ++r) {
        int row = m0 + r + half * 8;
        pre[(long)row * kG + col] = (bf16)(acc[c][r] + bias);
      }
    }
  }
}

// ---------------- persistent GRU recurrence (both directions of one sequence) ----------------
// 80 WGs = 2 streams x 8 M-tiles(64) x 5 H-tiles(64); grid barrier per timestep.
// The per-WG Whh slice (3 gates x 64 cols x 320 K bf16 = 120 KB) is staged in LDS once and
// reused for all 200 timesteps, keeping the serial critical path off L2.
__global__ void gru_kernel(const bf16* __restrict__ preF, const bf16* __restrict__ preB,
                           const bf16* __restrict__ WhhFp, const bf16* __restrict__ WhhBp,
                           const float* __restrict__ bhhF, const float* __restrict__ bhhB,
                           bf16* __restrict__ hbuf, bf16* __restrict__ o2, int* bar) {
  extern __shared__ char smem[];
  bf16* ldsW = (bf16*)smem;                  // [3*64][kKP]
  const int lane = threadIdx.x & 31, wave = threadIdx.x >> 5;
  const int half = lane >> 4, l16 = lane & 15;
  const int wg = blockIdx.x;                 // 0..79
  const int stream = wg / 40;                // 0=fwd, 1=bwd
  const int rem = wg % 40;
  const int m0 = (rem / 5) * 64 + wave * 16; // batch rows
  const int h0 = (rem % 5) * 64;             // hidden cols
  const bf16* Whh = stream ? WhhBp : WhhFp;
  const float* bhh = stream ? bhhB : bhhF;
  const bf16* pre = stream ? preB : preF;
  bf16* hs = hbuf + (long)stream * 2 * kB * kKP;

  // one-time LDS stage of the weight slice: rows {g*300 + h0 .. +63} for g=0..2
  for (int i = threadIdx.x; i < 3 * 64 * (kKP / 2); i += 128) {
    int row = i / (kKP / 2), c2 = i % (kKP / 2);
    int g = row >> 6, rl = row & 63;
    const unsigned int* src = (const unsigned int*)(Whh + (long)(g * kH + h0 + rl) * kKP);
    ((unsigned int*)ldsW)[i] = src[c2];
  }
  __syncthreads();

  for (int t = 0; t < kN; ++t) {
    const bf16* cur = hs + (long)(t & 1) * kB * kKP;
    bf16* nxt = hs + (long)((t + 1) & 1) * kB * kKP;
    int npos = stream ? (kN - 1 - t) : t;

    // hint the gate inputs into cache while the GEMM runs
    {
      long pb = ((long)(m0 + l16) * kN + npos) * kG + h0;
      __builtin_prefetch(pre + pb, 0, 0);
      __builtin_prefetch(pre + pb + kH, 0, 0);
      __builtin_prefetch(pre + pb + 2 * kH, 0, 0);
    }

    v8f acc[3][4];
#pragma unroll
    for (int g = 0; g < 3; ++g)
#pragma unroll
      for (int c = 0; c < 4; ++c) acc[g][c] = vzero();

    const bf16* arow = cur + (long)(m0 + l16) * kKP;
    for (int kt = 0; kt < 10; ++kt) {
      int k0 = kt * 32;
      v16bf a = frag_a(arow, k0, half);
#pragma unroll
      for (int g = 0; g < 3; ++g) {
#pragma unroll
        for (int c = 0; c < 4; ++c) {
          const bf16* brow = ldsW + (long)((g << 6) + c * 16 + l16) * kKP;
          acc[g][c] = WMMA_BF16(a, frag_b(brow, k0, half), acc[g][c]);
        }
      }
    }

#pragma unroll
    for (int c = 0; c < 4; ++c) {
      int col = h0 + c * 16 + l16;
      if (col < kH) {
        float bhr = bhh[col], bhz = bhh[col + kH], bhn = bhh[col + 2 * kH];
#pragma unroll
        for (int r = 0; r < 8; ++r) {
          int row = m0 + r + half * 8;
          long pi = ((long)row * kN + npos) * kG;
          float xr = (float)pre[pi + col];
          float xz = (float)pre[pi + col + kH];
          float xn = (float)pre[pi + col + 2 * kH];
          float hr = acc[0][c][r] + bhr;
          float hz = acc[1][c][r] + bhz;
          float hn = acc[2][c][r] + bhn;
          float rg = 1.f / (1.f + __expf(-(xr + hr)));
          float zg = 1.f / (1.f + __expf(-(xz + hz)));
          float ng = tanhf(xn + rg * hn);
          float hp = (float)cur[(long)row * kKP + col];
          float hv = (1.f - zg) * ng + zg * hp;
          nxt[(long)row * kKP + col] = (bf16)hv;
          o2[((long)row * kN + npos) * kO2W + stream * kH + col] = (bf16)hv;
        }
      }
    }
    grid_barrier(bar, 80);
  }
}

// ---------------- fused self-attention + mean/max pooling, one batch row per WG ----------------
// LDS: o2 tile 224x608 bf16 + 16x208 f32 score block + 16x224 bf16 attn block (~286 KB of 320 KB/WGP)
__global__ void attend_kernel(const bf16* __restrict__ o2, float* __restrict__ o8,
                              const float* __restrict__ sigma, int useBias) {
  extern __shared__ char smem[];
  bf16* o2t = (bf16*)smem;                                   // 224 x 608
  float* sblk = (float*)(smem + 224 * 608 * 2);              // 16 x 208
  bf16* ablk = (bf16*)(smem + 224 * 608 * 2 + 16 * 208 * 4); // 16 x 224
  const int tid = threadIdx.x;
  const int lane = tid & 31, wave = tid >> 5;
  const int half = lane >> 4, l16 = lane & 15;
  const int b = blockIdx.x;
  const float invSig = useBias ? (1.f / sigma[0]) : 0.f;

#if defined(__gfx1250__)
  // zero the whole padded tile, then DMA the valid 200x600 region straight into LDS via the
  // CDNA5 async-to-LDS path (ASYNCcnt-tracked, no VGPR round trip).
  for (int i = tid; i < 224 * 608 / 2; i += 256) ((unsigned int*)o2t)[i] = 0u;
  __syncthreads();
  {
    const char* gbase = (const char*)(o2 + (long)b * kN * kO2W);
    char* lbase = (char*)o2t;
    for (int i = tid; i < kN * 75; i += 256) {       // 75 x 16B chunks per 1200 B row
      int row = i / 75, c16 = i - row * 75;
      unsigned long long g =
          (unsigned long long)(gbase + (long)row * (kO2W * 2) + c16 * 16);
      unsigned int l =
          (unsigned int)(size_t)(lbase + (long)row * (608 * 2) + c16 * 16);
      // VDST = LDS byte address VGPR, VADDR = 64-bit global address (GV mode)
      asm volatile("global_load_async_to_lds_b128 %0, %1, off"
                   :: "v"(l), "v"(g) : "memory");
    }
  }
  asm volatile("s_wait_asynccnt 0" ::: "memory");
  __syncthreads();
#else
  for (int i = tid; i < 224 * 608; i += 256) {
    int r = i / 608, c = i - r * 608;
    bf16 v = (bf16)0.f;
    if (r < kN && c < kO2W) v = o2[((long)b * kN + r) * kO2W + c];
    o2t[i] = v;
  }
  __syncthreads();
#endif

  float colSum[5], colMax[5];
#pragma unroll
  for (int q = 0; q < 5; ++q) { colSum[q] = 0.f; colMax[q] = -3.4e38f; }

  for (int rb = 0; rb < 13; ++rb) {
    int r0 = rb * 16;
    // scores block: (16 x 208) = o2[r0:r0+16,:] @ o2^T, K = 608
    const bf16* arow = o2t + (r0 + l16) * 608;
    for (int tn = wave; tn < 13; tn += 8) {
      v8f acc = vzero();
      const bf16* brow = o2t + (tn * 16 + l16) * 608;
      for (int kt = 0; kt < 19; ++kt) {
        v16bf a = frag_a(arow, kt * 32, half);
        acc = WMMA_BF16(a, frag_b(brow, kt * 32, half), acc);
      }
      int colg = tn * 16 + l16;
#pragma unroll
      for (int r = 0; r < 8; ++r) {
        int rl = r + half * 8;
        int rowg = r0 + rl;
        float s = acc[r];
        if (useBias) { float d = (float)(rowg - colg); s -= d * d * invSig; }
        if (colg >= kN) s = -3.4e38f;
        sblk[rl * 208 + colg] = s;
      }
    }
    __syncthreads();
    // row softmax (one thread per row; 200 valid cols, padded cols are -inf -> 0)
    if (tid < 16) {
      float mx = -3.4e38f;
      for (int c = 0; c < 208; ++c) mx = fmaxf(mx, sblk[tid * 208 + c]);
      float sm = 0.f;
      for (int c = 0; c < 208; ++c) { float e = __expf(sblk[tid * 208 + c] - mx); sblk[tid * 208 + c] = e; sm += e; }
      float inv = 1.f / sm;
      for (int c = 0; c < 208; ++c) ablk[tid * 224 + c] = (bf16)(sblk[tid * 208 + c] * inv);
      for (int c = 208; c < 224; ++c) ablk[tid * 224 + c] = (bf16)0.f;
    }
    __syncthreads();
    // o5 block (16 x 600) = attn(16x224) @ o2(224x608), K padded to 224
    const bf16* aarow = ablk + l16 * 224;
    int q = 0;
    for (int tn = wave; tn < 38; tn += 8, ++q) {
      v8f acc = vzero();
      for (int kt = 0; kt < 7; ++kt) {
        v16bf a = frag_a(aarow, kt * 32, half);
        v16bf bb = frag_bT_lds(o2t + tn * 16 + l16, kt * 32, half, 608);
        acc = WMMA_BF16(a, bb, acc);
      }
#pragma unroll
      for (int r = 0; r < 8; ++r) {
        int rowg = r0 + r + half * 8;
        if (rowg < kN) {
          float v = acc[r];
          colSum[q] += v;
          colMax[q] = fmaxf(colMax[q], v);
        }
      }
    }
    __syncthreads();
  }
  // reduce lane pairs (L, L+16 share the same column) and emit mean/max
  int q = 0;
  for (int tn = wave; tn < 38; tn += 8, ++q) {
    float s = colSum[q], m = colMax[q];
    s += __shfl_xor(s, 16, 32);
    m = fmaxf(m, __shfl_xor(m, 16, 32));
    int col = tn * 16 + l16;
    if (half == 0 && col < kO2W) {
      o8[(long)b * 1200 + col] = s * (1.f / (float)kN);
      o8[(long)b * 1200 + 600 + col] = m;
    }
  }
}

// ---------------- head ----------------
__global__ void feat_kernel(const float* __restrict__ a, const float* __restrict__ bb,
                            bf16* __restrict__ f) {
  int total = kB * kFCK;
  for (int i = blockIdx.x * blockDim.x + threadIdx.x; i < total; i += gridDim.x * blockDim.x) {
    int b = i / kFCK, j = i - b * kFCK;
    int jj = (j < 1200) ? j : (j - 1200);
    float va = a[b * 1200 + jj], vb = bb[b * 1200 + jj];
    f[i] = (bf16)((j < 1200) ? fabsf(va - vb) : va * vb);
  }
}

// fc1: out[b][j] = relu(sum_k feat[b,k]*W[j,k] + bias[j]); M=512,N=512,K=2400
__global__ void fc1_kernel(const bf16* __restrict__ feat, const bf16* __restrict__ W,
                           const float* __restrict__ bias, bf16* __restrict__ out) {
  const int lane = threadIdx.x & 31, wave = threadIdx.x >> 5;
  const int half = lane >> 4, l16 = lane & 15;
  const int m0 = blockIdx.x * 64 + wave * 16;
  const int n0 = blockIdx.y * 64;

  v8f acc[4];
#pragma unroll
  for (int c = 0; c < 4; ++c) acc[c] = vzero();

  const bf16* arow = feat + (long)(m0 + l16) * kFCK;
  for (int kt = 0; kt < kFCK / 32; ++kt) {
    int k0 = kt * 32;
    v16bf a = frag_a(arow, k0, half);
#pragma unroll
    for (int c = 0; c < 4; ++c) {
      const bf16* wrow = W + (long)(n0 + c * 16 + l16) * kFCK;
      acc[c] = WMMA_BF16(a, frag_b(wrow, k0, half), acc[c]);
    }
  }
#pragma unroll
  for (int c = 0; c < 4; ++c) {
    int col = n0 + c * 16 + l16;
    float bv = bias[col];
#pragma unroll
    for (int r = 0; r < 8; ++r) {
      int row = m0 + r + half * 8;
      out[(long)row * kFCH + col] = (bf16)fmaxf(acc[c][r] + bv, 0.f);
    }
  }
}

__global__ void fc2_kernel(const bf16* __restrict__ h, const float* __restrict__ W,
                           const float* __restrict__ bias, float* __restrict__ out) {
  int i = blockIdx.x * blockDim.x + threadIdx.x;
  if (i < kB) {
    float s = 0.f;
    for (int k = 0; k < kFCH; ++k) s += (float)h[i * kFCH + k] * W[k];
    out[i] = 1.f / (1.f + __expf(-(s + bias[0])));
  }
}

extern "C" void kernel_launch(void* const* d_in, const int* in_sizes, int n_in,
                              void* d_out, int out_size, void* d_ws, size_t ws_size,
                              hipStream_t stream) {
  (void)in_sizes; (void)n_in; (void)out_size; (void)ws_size;
  const int*   x      = (const int*)d_in[0];
  const float* embed  = (const float*)d_in[1];
  const float* Wih_f  = (const float*)d_in[2];
  const float* Whh_f  = (const float*)d_in[3];
  const float* bih_f  = (const float*)d_in[4];
  const float* bhh_f  = (const float*)d_in[5];
  const float* Wih_b  = (const float*)d_in[6];
  const float* Whh_b  = (const float*)d_in[7];
  const float* bih_b  = (const float*)d_in[8];
  const float* bhh_b  = (const float*)d_in[9];
  const float* sigma  = (const float*)d_in[10];
  const float* fc1_W  = (const float*)d_in[11];
  const float* fc1_b  = (const float*)d_in[12];
  const float* fc2_W  = (const float*)d_in[13];
  const float* fc2_b  = (const float*)d_in[14];
  float* out = (float*)d_out;

  char* ws = (char*)d_ws;
  size_t off = 0;
  auto carve = [&](size_t bytes) -> void* {
    void* p = ws + off;
    off += (bytes + 255) & ~(size_t)255;
    return p;
  };
  bf16* embBf = (bf16*)carve((size_t)kVoc * kKP * 2);        // 32 MB padded bf16 table
  bf16* WihFp = (bf16*)carve((size_t)kRP * kKP * 2);
  bf16* WihBp = (bf16*)carve((size_t)kRP * kKP * 2);
  bf16* WhhFp = (bf16*)carve((size_t)kRP * kKP * 2);
  bf16* WhhBp = (bf16*)carve((size_t)kRP * kKP * 2);
  bf16* fc1Wb = (bf16*)carve((size_t)kFCH * kFCK * 2);
  bf16* preF  = (bf16*)carve((size_t)kB * kN * kG * 2);
  bf16* preB  = (bf16*)carve((size_t)kB * kN * kG * 2);
  bf16* o2b   = (bf16*)carve((size_t)kB * kN * kO2W * 2);
  bf16* hbuf  = (bf16*)carve((size_t)2 * 2 * kB * kKP * 2);  // streams x ping-pong
  int*  bar   = (int*)carve(256);                            // directly after hbuf
  float* o8a  = (float*)carve((size_t)kB * 1200 * 4);
  float* o8b  = (float*)carve((size_t)kB * 1200 * 4);
  bf16* featb = (bf16*)carve((size_t)kB * kFCK * 2);
  bf16* fc1o  = (bf16*)carve((size_t)kB * kFCH * 2);

  // conversions (bf16, zero-padded K and rows)
  cvt_pad_kernel<<<4096, 256, 0, stream>>>(embed, embBf, kVoc, kV, kVoc, kKP);
  cvt_pad_kernel<<<512, 256, 0, stream>>>(Wih_f, WihFp, kG, kV, kRP, kKP);
  cvt_pad_kernel<<<512, 256, 0, stream>>>(Wih_b, WihBp, kG, kV, kRP, kKP);
  cvt_pad_kernel<<<512, 256, 0, stream>>>(Whh_f, WhhFp, kG, kH, kRP, kKP);
  cvt_pad_kernel<<<512, 256, 0, stream>>>(Whh_b, WhhBp, kG, kH, kRP, kKP);
  cvt_pad_kernel<<<512, 256, 0, stream>>>(fc1_W, fc1Wb, kFCH, kFCK, kFCH, kFCK);

  const size_t attnLds = (size_t)224 * 608 * 2 + (size_t)16 * 208 * 4 + (size_t)16 * 224 * 2;
  const size_t gruLds  = (size_t)3 * 64 * kKP * 2;   // 120 KB Whh slice

  for (int s = 0; s < 2; ++s) {
    proj_kernel<<<dim3((kB * kN) / 64, 15, 2), 128, 0, stream>>>(
        x, embBf, WihFp, WihBp, bih_f, bih_b, preF, preB, s);
    // zero h ping-pong buffers and the grid barrier (contiguous in ws)
    zero_kernel<<<512, 256, 0, stream>>>(
        (unsigned int*)hbuf, (long)((2 * 2 * kB * kKP * 2 + 256) / 4));
    gru_kernel<<<80, 128, gruLds, stream>>>(preF, preB, WhhFp, WhhBp, bhh_f, bhh_b, hbuf, o2b, bar);
    attend_kernel<<<kB, 256, attnLds, stream>>>(o2b, s ? o8b : o8a, sigma, s ? 1 : 0);
  }

  feat_kernel<<<4800, 256, 0, stream>>>(o8a, o8b, featb);
  fc1_kernel<<<dim3(kB / 64, kFCH / 64), 128, 0, stream>>>(featb, fc1Wb, fc1_b, fc1o);
  fc2_kernel<<<2, 256, 0, stream>>>(fc1o, fc2_W, fc2_b, out);
}
